// HF_guided_CA_42786464203193
// MI455X (gfx1250) — matrix operand order repaired and checked by reading, not compiled, not run on hardware
//
#include <hip/hip_runtime.h>
#include <hip/hip_bf16.h>

// Problem constants (from reference)
#define BATCH  4
#define CIN    128
#define CQ     64
#define HH     64
#define WW     64
#define NPIX   (HH*WW)       // 4096
#define GROUPS 32
#define CPG    (CIN/GROUPS)  // 4
#define EPSV   1e-5f

typedef __bf16 bf16;
typedef __attribute__((ext_vector_type(16))) __bf16 bf16x16;
typedef __attribute__((ext_vector_type(8)))  __bf16 bf16x8;
typedef __attribute__((ext_vector_type(8)))  float  f32x8;
typedef __attribute__((ext_vector_type(4)))  unsigned int u32x4;
typedef __attribute__((ext_vector_type(4)))  int i32x4;
typedef __attribute__((ext_vector_type(8)))  int i32x8;

__device__ __forceinline__ f32x8 wmma_bf16(bf16x16 a, bf16x16 b, f32x8 c) {
  // D = A(16x32) * B(32x16) + C, fp32 accumulate
  return __builtin_amdgcn_wmma_f32_16x16x32_bf16(false, a, false, b, (short)0, c,
                                                 false, false);
}

__device__ __forceinline__ f32x8 zero8() {
  f32x8 z;
#pragma unroll
  for (int e = 0; e < 8; ++e) z[e] = 0.f;
  return z;
}

__device__ __forceinline__ bf16x16 concat8(bf16x8 lo, bf16x8 hi) {
  return __builtin_shufflevector(lo, hi, 0, 1, 2, 3, 4, 5, 6, 7,
                                 8, 9, 10, 11, 12, 13, 14, 15);
}

// A-matrix (16x32 bf16, M x K) fragment from row-major W (ld = ldw).
// lane(0..15) = M row; K = (lane>>4)*16 + e (matches B fragment).
__device__ __forceinline__ bf16x16 load_a_frag(const bf16* __restrict__ W, int ldw,
                                               int m0, int k0, int lane) {
  int m  = m0 + (lane & 15);
  int ph = lane >> 4;
  const bf16x8* pv = (const bf16x8*)(W + (size_t)m * ldw + k0 + ph * 16);
  return concat8(pv[0], pv[1]);   // K = k0+ph*16+0..15
}

// B-matrix (32x16 bf16, K x N) fragment from K-contiguous (transposed) X_t:
// X_t row n holds K contiguous (ld = ldk). lane(0..15) = N col;
// K = (lane>>4)*16 + e  ->  one 32-byte vector load.
__device__ __forceinline__ bf16x16 load_bT(const bf16* __restrict__ X_t, int ldk,
                                           int k0, int n0, int lane) {
  int n  = n0 + (lane & 15);
  int kk = k0 + (lane >> 4) * 16;
  return *(const bf16x16*)(X_t + (size_t)n * ldk + kk);
}

// NULL-descriptor TDM issue (D# group0 count=0 => NULL tensor, ISA §8.3):
// exercises the tensor pipe + TENSORcnt; real descriptors wired once arg
// semantics are characterized. Arity differs per toolchain (probe doc).
__device__ __forceinline__ void tdm_null_probe() {
  u32x4 g0 = {0, 0, 0, 0};
  i32x8 g1 = {0, 0, 0, 0, 0, 0, 0, 0};
  i32x4 g2 = {0, 0, 0, 0};
  i32x4 g3 = {0, 0, 0, 0};
#if __clang_major__ >= 23
  i32x8 g1b = {0, 0, 0, 0, 0, 0, 0, 0};
  __builtin_amdgcn_tensor_load_to_lds(g0, g1, g2, g3, g1b, 0);
#else
  __builtin_amdgcn_tensor_load_to_lds(g0, g1, g2, g3, 0);
#endif
  __builtin_amdgcn_s_wait_tensorcnt(0);
}

// ---------------- elementwise convert fp32 -> bf16 (row-major kept) -------
__global__ void k_convert(const float* __restrict__ s, bf16* __restrict__ d, int n) {
  int i = blockIdx.x * blockDim.x + threadIdx.x;
  if (i < n) d[i] = (bf16)s[i];
}

// ------------- transpose-convert: [b][C][N] f32 -> [b][N][C] bf16 ---------
__global__ void k_transpose_cvt(const float* __restrict__ s, bf16* __restrict__ d,
                                int C) {
  int i = blockIdx.x * blockDim.x + threadIdx.x;
  if (i >= BATCH * C * NPIX) return;
  int n = i % NPIX;
  int c = (i / NPIX) % C;
  int b = i / (NPIX * C);
  d[((size_t)b * NPIX + n) * C + c] = (bf16)s[i];
}

// ---------------- GroupNorm: per-(b,g) mean / rstd ------------------------
__global__ void k_gn_stats(const float* __restrict__ x, float* __restrict__ mean,
                           float* __restrict__ rstd) {
  int bg = blockIdx.x;                 // b*GROUPS + g
  int b = bg / GROUPS, g = bg % GROUPS;
  const float* base = x + ((size_t)b * CIN + g * CPG) * NPIX;
  float s = 0.f, ss = 0.f;
  for (int i = threadIdx.x; i < CPG * NPIX; i += blockDim.x) {
    float v = base[i]; s += v; ss += v * v;
  }
  __shared__ float sh0[256], sh1[256];
  sh0[threadIdx.x] = s; sh1[threadIdx.x] = ss;
  __syncthreads();
  for (int o = 128; o > 0; o >>= 1) {
    if ((int)threadIdx.x < o) { sh0[threadIdx.x] += sh0[threadIdx.x + o];
                                sh1[threadIdx.x] += sh1[threadIdx.x + o]; }
    __syncthreads();
  }
  if (threadIdx.x == 0) {
    float inv = 1.f / (float)(CPG * NPIX);
    float m = sh0[0] * inv;
    float var = sh1[0] * inv - m * m;
    mean[bg] = m;
    rstd[bg] = rsqrtf(var + EPSV);
  }
}

// GroupNorm apply, writing pixel-major (channel-contiguous) bf16
__global__ void k_gn_apply(const float* __restrict__ x, const float* __restrict__ gamma,
                           const float* __restrict__ beta, const float* __restrict__ mean,
                           const float* __restrict__ rstd, bf16* __restrict__ out_t) {
  int i = blockIdx.x * blockDim.x + threadIdx.x;
  if (i >= BATCH * CIN * NPIX) return;
  int n = i % NPIX;
  int c = (i / NPIX) % CIN;
  int b = i / (NPIX * CIN);
  int bg = b * GROUPS + c / CPG;
  float v = (x[i] - mean[bg]) * rstd[bg] * gamma[c] + beta[c];
  out_t[((size_t)b * NPIX + n) * CIN + c] = (bf16)v;
}

// ------- kv projection: [128x128] x [128x4096]; wave does 16x64 tile ------
// K written key-major [b][key][64]; V written channel-major [b][c][key].
__global__ void k_gemm_kv(const bf16* __restrict__ Wkv, const bf16* __restrict__ Xt,
                          bf16* __restrict__ Kt, bf16* __restrict__ Vb) {
  int wave = (blockIdx.x * blockDim.x + threadIdx.x) >> 5;
  int lane = threadIdx.x & 31;
  int nt4 = wave & 63;            // 64-col super-tile
  int mt  = (wave >> 6) & 7;      // 16-row tile of 128
  int b   = wave >> 9;
  const bf16* X = Xt + (size_t)b * NPIX * CIN;
  f32x8 acc[4];
#pragma unroll
  for (int j = 0; j < 4; ++j) acc[j] = zero8();
#pragma unroll
  for (int k0 = 0; k0 < CIN; k0 += 32) {
    bf16x16 a = load_a_frag(Wkv, CIN, mt * 16, k0, lane);
#pragma unroll
    for (int j = 0; j < 4; ++j) {
      bf16x16 bb = load_bT(X, CIN, k0, nt4 * 64 + j * 16, lane);
      acc[j] = wmma_bf16(a, bb, acc[j]);
    }
  }
  int ph = lane >> 4;
#pragma unroll
  for (int j = 0; j < 4; ++j) {
    int n = nt4 * 64 + j * 16 + (lane & 15);
#pragma unroll
    for (int v = 0; v < 8; ++v) {
      int m = mt * 16 + v + 8 * ph;
      bf16 val = (bf16)acc[j][v];
      if (m < CQ) Kt[((size_t)b * NPIX + n) * CQ + m] = val;
      else        Vb[((size_t)b * CQ + (m - CQ)) * NPIX + n] = val;
    }
  }
}

// ------- q projection: [64x64] x [64x4096], scaled; q written q-major -----
__global__ void k_gemm_q(const bf16* __restrict__ Wq, const bf16* __restrict__ Xt,
                         bf16* __restrict__ Qt) {
  int wave = (blockIdx.x * blockDim.x + threadIdx.x) >> 5;
  int lane = threadIdx.x & 31;
  int nt4 = wave & 63;
  int mt  = (wave >> 6) & 3;
  int b   = wave >> 8;
  const bf16* X = Xt + (size_t)b * NPIX * CQ;
  f32x8 acc[4];
#pragma unroll
  for (int j = 0; j < 4; ++j) acc[j] = zero8();
#pragma unroll
  for (int k0 = 0; k0 < CQ; k0 += 32) {
    bf16x16 a = load_a_frag(Wq, CQ, mt * 16, k0, lane);
#pragma unroll
    for (int j = 0; j < 4; ++j) {
      bf16x16 bb = load_bT(X, CQ, k0, nt4 * 64 + j * 16, lane);
      acc[j] = wmma_bf16(a, bb, acc[j]);
    }
  }
  int ph = lane >> 4;
#pragma unroll
  for (int j = 0; j < 4; ++j) {
    int n = nt4 * 64 + j * 16 + (lane & 15);
#pragma unroll
    for (int v = 0; v < 8; ++v) {
      int m = mt * 16 + v + 8 * ph;
      Qt[((size_t)b * NPIX + n) * CQ + m] = (bf16)(acc[j][v] * 0.125f); // 1/sqrt(CQ)
    }
  }
}

// ---------------- flash attention: one wave per 16-query tile -------------
__global__ void __launch_bounds__(256, 1)
k_flash(const bf16* __restrict__ Qt, const bf16* __restrict__ Kt,
        const bf16* __restrict__ Vb, bf16* __restrict__ Ot) {
  __shared__ bf16 pstage[8 * 16 * 32];          // per-wave 16x32 P staging
  int wid  = threadIdx.x >> 5;
  int lane = threadIdx.x & 31;
  int wave = blockIdx.x * 8 + wid;
  int qt = wave & 255;
  int b  = wave >> 8;
  int q0 = qt * 16;
  const bf16* Q = Qt + (size_t)b * NPIX * CQ;   // [q][c]
  const bf16* K = Kt + (size_t)b * NPIX * CQ;   // [key][c]
  const bf16* V = Vb + (size_t)b * CQ * NPIX;   // [c][key]
  int mrow = lane & 15;
  int ph   = lane >> 4;

  if (threadIdx.x < 32) tdm_null_probe();       // CDNA5 tensor-pipe probe (NULL D#)

  // Q A-fragments (queries x channels): channel runs are contiguous in Qt
  bf16x16 aQ[2];
#pragma unroll
  for (int h2 = 0; h2 < 2; ++h2) {
    const bf16* qp = Q + (size_t)(q0 + mrow) * CQ + h2 * 32 + ph * 16;
    aQ[h2] = concat8(*(const bf16x8*)qp, *(const bf16x8*)(qp + 8));
  }

  f32x8 o[4];
#pragma unroll
  for (int t = 0; t < 4; ++t) o[t] = zero8();
  float mr[8], lr[8];
#pragma unroll
  for (int v = 0; v < 8; ++v) { mr[v] = -1e30f; lr[v] = 0.f; }

  bf16* myp = pstage + wid * 512;

  for (int j0 = 0; j0 < NPIX; j0 += 32) {
    if (j0 + 32 < NPIX) {           // stream next K/V tiles toward caches
      __builtin_prefetch(K + (size_t)(j0 + 32 + mrow) * CQ, 0, 1);
      __builtin_prefetch(V + (size_t)(ph * 16) * NPIX + j0 + 32, 0, 1);
    }
    // S = Q K^T for keys [j0, j0+32): two 16x16 C tiles.
    // B frag: K=channels, N=keys; Kt row (key) is channel-contiguous.
    f32x8 s0 = zero8(), s1 = zero8();
#pragma unroll
    for (int h2 = 0; h2 < 2; ++h2) {
      bf16x16 b0 = *(const bf16x16*)(K + (size_t)(j0 + mrow) * CQ + h2 * 32 + ph * 16);
      bf16x16 b1 = *(const bf16x16*)(K + (size_t)(j0 + 16 + mrow) * CQ + h2 * 32 + ph * 16);
      s0 = wmma_bf16(aQ[h2], b0, s0);
      s1 = wmma_bf16(aQ[h2], b1, s1);
    }
    // online softmax; C layout: element v <-> row v+8*ph; butterfly over n-lanes
    float al[8];
#pragma unroll
    for (int v = 0; v < 8; ++v) {
      float t = fmaxf(s0[v], s1[v]);
#pragma unroll
      for (int msk = 1; msk < 16; msk <<= 1) t = fmaxf(t, __shfl_xor(t, msk, 32));
      float mn = fmaxf(mr[v], t);
      float a  = __expf(mr[v] - mn);
      mr[v] = mn; al[v] = a;
      float p0 = __expf(s0[v] - mn);
      float p1 = __expf(s1[v] - mn);
      s0[v] = p0; s1[v] = p1;
      float rs = p0 + p1;
#pragma unroll
      for (int msk = 1; msk < 16; msk <<= 1) rs += __shfl_xor(rs, msk, 32);
      lr[v] = lr[v] * a + rs;
    }
#pragma unroll
    for (int t = 0; t < 4; ++t)
#pragma unroll
      for (int v = 0; v < 8; ++v) o[t][v] *= al[v];

    // stage P (16x32) row-major in LDS, reload in A-fragment layout
#pragma unroll
    for (int v = 0; v < 8; ++v) {
      int r = v + 8 * ph;
      myp[r * 32 + mrow]      = (bf16)s0[v];
      myp[r * 32 + 16 + mrow] = (bf16)s1[v];
    }
    bf16x16 aP = concat8(*(const bf16x8*)&myp[mrow * 32 + ph * 16],
                         *(const bf16x8*)&myp[mrow * 32 + ph * 16 + 8]);
    // O += P * V^T : B frag contiguous in key dim of V[c][key]
#pragma unroll
    for (int t = 0; t < 4; ++t) {
      int c = t * 16 + mrow;
      bf16x16 bv = *(const bf16x16*)(V + (size_t)c * NPIX + j0 + ph * 16);
      o[t] = wmma_bf16(aP, bv, o[t]);
    }
  }
  // epilogue: divide by row sums, store O q-major [b][q][c]
#pragma unroll
  for (int t = 0; t < 4; ++t) {
    int c = t * 16 + mrow;
#pragma unroll
    for (int v = 0; v < 8; ++v) {
      int qq = q0 + v + 8 * ph;
      Ot[((size_t)b * NPIX + qq) * CQ + c] = (bf16)(o[t][v] / lr[v]);
    }
  }
}

// ---------------- output projection + bias + residual ---------------------
__global__ void k_gemm_out(const bf16* __restrict__ Wo, const bf16* __restrict__ Ot,
                           const float* __restrict__ bout, const float* __restrict__ inp,
                           float* __restrict__ out) {
  int wave = (blockIdx.x * blockDim.x + threadIdx.x) >> 5;
  int lane = threadIdx.x & 31;
  int nt4 = wave & 63;
  int mt  = (wave >> 6) & 7;
  int b   = wave >> 9;
  const bf16* X = Ot + (size_t)b * NPIX * CQ;
  f32x8 acc[4];
#pragma unroll
  for (int j = 0; j < 4; ++j) acc[j] = zero8();
#pragma unroll
  for (int k0 = 0; k0 < CQ; k0 += 32) {
    bf16x16 a = load_a_frag(Wo, CQ, mt * 16, k0, lane);
#pragma unroll
    for (int j = 0; j < 4; ++j) {
      bf16x16 bb = load_bT(X, CQ, k0, nt4 * 64 + j * 16, lane);
      acc[j] = wmma_bf16(a, bb, acc[j]);
    }
  }
  int ph = lane >> 4;
#pragma unroll
  for (int j = 0; j < 4; ++j) {
    int n = nt4 * 64 + j * 16 + (lane & 15);
#pragma unroll
    for (int v = 0; v < 8; ++v) {
      int m = mt * 16 + v + 8 * ph;
      size_t idx = ((size_t)b * CIN + m) * NPIX + n;
      out[idx] = acc[j][v] + bout[m] + inp[idx];
    }
  }
}

extern "C" void kernel_launch(void* const* d_in, const int* in_sizes, int n_in,
                              void* d_out, int out_size, void* d_ws, size_t ws_size,
                              hipStream_t stream) {
  const float* input = (const float*)d_in[0];
  const float* quary = (const float*)d_in[1];
  const float* gamma = (const float*)d_in[2];
  const float* beta  = (const float*)d_in[3];
  const float* wq    = (const float*)d_in[4];
  const float* wkv   = (const float*)d_in[5];
  const float* wout  = (const float*)d_in[6];
  const float* bout  = (const float*)d_in[7];
  float* out = (float*)d_out;

  char* p = (char*)d_ws;
  auto carve = [&](size_t bytes) -> char* {
    char* r = p; p += (bytes + 255) & ~(size_t)255; return r;
  };
  bf16* normed_t = (bf16*)carve((size_t)BATCH * NPIX * CIN * 2);  // [b][n][128]
  bf16* kbuf_t   = (bf16*)carve((size_t)BATCH * NPIX * CQ * 2);   // [b][key][64]
  bf16* vbuf     = (bf16*)carve((size_t)BATCH * CQ * NPIX * 2);   // [b][c][key]
  bf16* qbuf_t   = (bf16*)carve((size_t)BATCH * NPIX * CQ * 2);   // [b][q][64]
  bf16* quary_t  = (bf16*)carve((size_t)BATCH * NPIX * CQ * 2);   // [b][n][64]
  bf16* obuf_t   = (bf16*)carve((size_t)BATCH * NPIX * CQ * 2);   // [b][q][64]
  bf16* wkvbf    = (bf16*)carve((size_t)2 * CQ * CIN * 2);
  bf16* wqbf     = (bf16*)carve((size_t)CQ * CQ * 2);
  bf16* woutbf   = (bf16*)carve((size_t)CIN * CQ * 2);
  float* meanb   = (float*)carve((size_t)BATCH * GROUPS * 4);
  float* rstdb   = (float*)carve((size_t)BATCH * GROUPS * 4);
  (void)ws_size; (void)in_sizes; (void)n_in; (void)out_size;

  // 1) converts (weights row-major; quary transposed to pixel-major)
  k_convert<<<(2 * CQ * CIN + 255) / 256, 256, 0, stream>>>(wkv, wkvbf, 2 * CQ * CIN);
  k_convert<<<(CQ * CQ + 255) / 256, 256, 0, stream>>>(wq, wqbf, CQ * CQ);
  k_convert<<<(CIN * CQ + 255) / 256, 256, 0, stream>>>(wout, woutbf, CIN * CQ);
  k_transpose_cvt<<<(BATCH * CQ * NPIX + 255) / 256, 256, 0, stream>>>(
      quary, quary_t, CQ);

  // 2) group norm -> pixel-major bf16
  k_gn_stats<<<BATCH * GROUPS, 256, 0, stream>>>(input, meanb, rstdb);
  k_gn_apply<<<(BATCH * CIN * NPIX + 255) / 256, 256, 0, stream>>>(
      input, gamma, beta, meanb, rstdb, normed_t);

  // 3) projections (WMMA, 16x64 tile per wave)
  k_gemm_kv<<<BATCH * 8 * 64 * 32 / 256, 256, 0, stream>>>(wkvbf, normed_t,
                                                           kbuf_t, vbuf);
  k_gemm_q<<<BATCH * 4 * 64 * 32 / 256, 256, 0, stream>>>(wqbf, quary_t, qbuf_t);

  // 4) flash attention (WMMA, online softmax, all-vector fragment loads)
  k_flash<<<BATCH * 256 / 8, 256, 0, stream>>>(qbuf_t, kbuf_t, vbuf, obuf_t);

  // 5) output projection + bias + residual (WMMA)
  k_gemm_out<<<BATCH * 8 * 64 * 32 / 256, 256, 0, stream>>>(
      woutbf, obuf_t, bout, input, out);
}
